// Decoder_45689862095084
// MI455X (gfx1250) — matrix operand order repaired and checked by reading, not compiled, not run on hardware
//
#include <hip/hip_runtime.h>
#include <hip/hip_bf16.h>
#include <math.h>

// Problem constants (match reference)
#define B_  4
#define L_  2048
#define D_  768
#define H_  12
#define I_  3072
#define HD_ 64
#define M_  (B_ * L_)   // 8192 token rows

typedef __bf16 bf16;
typedef __attribute__((ext_vector_type(16))) __bf16 v16bf;
typedef __attribute__((ext_vector_type(8)))  __bf16 v8bf;
typedef __attribute__((ext_vector_type(8)))  float  v8f;

union FragU {
    v16bf v;
    v8bf  h[2];
    __bf16 e[16];
};

__device__ __forceinline__ v8f wmma_bf16(v16bf a, v16bf b, v8f c) {
    // D = A(16x32 bf16) * B(32x16 bf16) + C(16x16 f32)
    return __builtin_amdgcn_wmma_f32_16x16x32_bf16(
        /*neg_a=*/false, a, /*neg_b=*/false, b,
        /*c_mod=*/(short)0, c, /*reuse_a=*/false, /*reuse_b=*/false);
}

// ---------------------------------------------------------------------------
// Weight convert + transpose: out[n*K + k] = (bf16) in[k*N + n]
// ---------------------------------------------------------------------------
__global__ void transpose_to_bf16(const float* __restrict__ in,
                                  bf16* __restrict__ out, int K, int N) {
    size_t idx = (size_t)blockIdx.x * 256 + threadIdx.x;
    size_t total = (size_t)K * N;
    if (idx >= total) return;
    int k = (int)(idx / N);
    int n = (int)(idx % N);
    out[(size_t)n * K + k] = (bf16)in[idx];
}

// ---------------------------------------------------------------------------
// LayerNorm over D=768, one block (256 threads) per row; bf16 output
// ---------------------------------------------------------------------------
__global__ void ln_bf16_kernel(const float* __restrict__ in,
                               const float* __restrict__ g,
                               const float* __restrict__ bvec,
                               bf16* __restrict__ out) {
    __shared__ float s_sum[256];
    __shared__ float s_sq[256];
    const int row = blockIdx.x;
    const float* x = in + (size_t)row * D_;
    float v[3];
    float sum = 0.f, sq = 0.f;
#pragma unroll
    for (int i = 0; i < 3; ++i) {
        v[i] = x[threadIdx.x + i * 256];
        sum += v[i];
        sq  += v[i] * v[i];
    }
    s_sum[threadIdx.x] = sum;
    s_sq[threadIdx.x]  = sq;
    __syncthreads();
    for (int off = 128; off > 0; off >>= 1) {
        if (threadIdx.x < off) {
            s_sum[threadIdx.x] += s_sum[threadIdx.x + off];
            s_sq[threadIdx.x]  += s_sq[threadIdx.x + off];
        }
        __syncthreads();
    }
    const float mu  = s_sum[0] * (1.0f / D_);
    const float var = s_sq[0] * (1.0f / D_) - mu * mu;
    const float rs  = rsqrtf(var + 1e-5f);
    bf16* o = out + (size_t)row * D_;
#pragma unroll
    for (int i = 0; i < 3; ++i) {
        int c = threadIdx.x + i * 256;
        o[c] = (bf16)((v[i] - mu) * rs * g[c] + bvec[c]);
    }
}

// ---------------------------------------------------------------------------
// WMMA GEMM: out[M,N] = A[M,K](bf16) * WT[N,K](bf16, pre-transposed) + bias
// Optional: residual add (f32), exact GELU, f32 and/or bf16 store, and
// V-transpose scatter (vT[b][h][d][l]) for the QKV GEMM.
// Block = 256 threads = 8 waves; wave tile 32x64; block tile 64(M) x 256(N).
// ---------------------------------------------------------------------------
__global__ void gemm_wmma(const bf16* __restrict__ A,
                          const bf16* __restrict__ WT,
                          const float* __restrict__ bias,
                          const float* __restrict__ residual,
                          float* __restrict__ outF,
                          bf16* __restrict__ outB,
                          bf16* __restrict__ vT,
                          int M, int N, int K, int gelu) {
    const int lane = threadIdx.x & 31;
    const int wid  = threadIdx.x >> 5;
    const int nIdx = lane & 15;
    const int half = lane >> 4;
    const int m0 = blockIdx.y * 64  + (wid & 1) * 32;
    const int n0 = blockIdx.x * 256 + (wid >> 1) * 64;

    v8f c[2][4] = {};

    for (int kb = 0; kb < K; kb += 32) {
        FragU a[2], bfr[4];
        // A fragments (lane = M row; K = kb + 16*half + s, contiguous per lane-half)
#pragma unroll
        for (int mi = 0; mi < 2; ++mi) {
            const bf16* p = A + (size_t)(m0 + mi * 16 + nIdx) * K + kb + half * 16;
            a[mi].h[0] = *(const v8bf*)p;          // K = kb + half*16 .. +7
            a[mi].h[1] = *(const v8bf*)(p + 8);    // K = kb + half*16 + 8 .. +15
        }
        // B fragments from transposed weights (lane = N col; K = kb + 16*half + j)
#pragma unroll
        for (int ni = 0; ni < 4; ++ni) {
            const bf16* p = WT + (size_t)(n0 + ni * 16 + nIdx) * K + kb + half * 16;
            bfr[ni].h[0] = *(const v8bf*)p;
            bfr[ni].h[1] = *(const v8bf*)(p + 8);
            __builtin_prefetch((const void*)(p + 32), 0, 1);
        }
#pragma unroll
        for (int mi = 0; mi < 2; ++mi)
#pragma unroll
            for (int ni = 0; ni < 4; ++ni)
                c[mi][ni] = wmma_bf16(a[mi].v, bfr[ni].v, c[mi][ni]);
    }

    // Epilogue: C/D layout row = r + 8*half, col = nIdx
#pragma unroll
    for (int mi = 0; mi < 2; ++mi)
#pragma unroll
        for (int ni = 0; ni < 4; ++ni)
#pragma unroll
            for (int r = 0; r < 8; ++r) {
                const int m = m0 + mi * 16 + r + 8 * half;
                const int n = n0 + ni * 16 + nIdx;
                const size_t o = (size_t)m * N + n;
                float v = c[mi][ni][r] + bias[n];
                if (residual) v += residual[o];
                if (gelu) v = 0.5f * v * (1.0f + erff(v * 0.70710678118654752f));
                if (outF) outF[o] = v;
                if (outB) outB[o] = (bf16)v;
                if (vT) {   // QKV GEMM: scatter V columns into [B,H,HD,L]
                    const int inner = n % 192;
                    if (inner >= 128) {
                        const int hh = n / 192;
                        const int bb = m >> 11;        // m / L_
                        const int ll = m & (L_ - 1);   // m % L_
                        vT[(((size_t)bb * H_ + hh) * HD_ + (inner - 128)) * L_ + ll] = (bf16)v;
                    }
                }
            }
}

// ---------------------------------------------------------------------------
// Flash attention (causal + padding mask). One wave per 16 query rows per
// (b,h). qkv layout: [B*L, H*192] with q at h*192, k at +64.
// V comes from vT: bf16 [B,H,HD,L] so P*V B-fragments are contiguous loads.
// Output attnout: bf16 [B*L, 768] with col = h*64 + d.
// ---------------------------------------------------------------------------
__global__ void attn_kernel(const bf16* __restrict__ qkv,
                            const bf16* __restrict__ vT,
                            const unsigned char* __restrict__ amask,
                            bf16* __restrict__ attnout) {
    __shared__ __align__(16) bf16 Pl[8][16 * 32];   // per-wave P tile (16x32)

    const int lane = threadIdx.x & 31;
    const int wid  = threadIdx.x >> 5;
    const int nIdx = lane & 15;
    const int half = lane >> 4;

    const int TPB  = L_ / 16;                        // 128 row tiles per (b,h)
    const int tile = blockIdx.x * 8 + wid;
    const int b    = tile / (H_ * TPB);
    const int rem  = tile % (H_ * TPB);
    const int h    = rem / TPB;
    const int rb   = (rem % TPB) * 16;
    const size_t baseBL = (size_t)b * L_;

    // Q fragments: constant across the key loop (A layout: lane = M = query row)
    FragU aQ[2];
    {
        const bf16* qrow = qkv + (baseBL + rb + nIdx) * 2304 + h * 192;
#pragma unroll
        for (int c2 = 0; c2 < 2; ++c2) {
            const bf16* p = qrow + c2 * 32 + half * 16;
            aQ[c2].h[0] = *(const v8bf*)p;
            aQ[c2].h[1] = *(const v8bf*)(p + 8);
        }
    }

    float mrow[8], lrow[8];
    v8f acc[4] = {};
#pragma unroll
    for (int r = 0; r < 8; ++r) { mrow[r] = -1e30f; lrow[r] = 0.0f; }

    const float scale = 1.0f / (float)HD_;
    const int kend = rb + 16;                        // causal: keys < rb+16
    const bf16* vbase = vT + ((size_t)b * H_ + h) * HD_ * L_;

    for (int jb = 0; jb < kend; jb += 32) {
        // ---- S = Q K^T for 32 keys (two 16-wide tiles) ----
        v8f s0 = {}, s1 = {};
        {
            const int key0 = jb + nIdx;        // <= 2031, in range
            const int key1 = jb + 16 + nIdx;   // <= 2047, in range
            const bf16* k0 = qkv + (baseBL + key0) * 2304 + h * 192 + 64 + half * 16;
            const bf16* k1 = qkv + (baseBL + key1) * 2304 + h * 192 + 64 + half * 16;
#pragma unroll
            for (int c2 = 0; c2 < 2; ++c2) {
                FragU bk0, bk1;
                bk0.h[0] = *(const v8bf*)(k0 + c2 * 32);
                bk0.h[1] = *(const v8bf*)(k0 + c2 * 32 + 8);
                bk1.h[0] = *(const v8bf*)(k1 + c2 * 32);
                bk1.h[1] = *(const v8bf*)(k1 + c2 * 32 + 8);
                s0 = wmma_bf16(aQ[c2].v, bk0.v, s0);
                s1 = wmma_bf16(aQ[c2].v, bk1.v, s1);
            }
        }

        // ---- online softmax (per output row r + 8*half) ----
#pragma unroll
        for (int r = 0; r < 8; ++r) {
            const int row  = rb + r + 8 * half;
            const int key0 = jb + nIdx;
            const int key1 = key0 + 16;
            float v0 = (key0 <= row && amask[b * L_ + key0]) ? s0[r] * scale : -1e30f;
            float v1 = (key1 <= row && amask[b * L_ + key1]) ? s1[r] * scale : -1e30f;
            float mx = fmaxf(v0, v1);
#pragma unroll
            for (int off = 8; off > 0; off >>= 1)
                mx = fmaxf(mx, __shfl_xor(mx, off, 32));    // reduce within 16-lane half
            const float mnew  = fmaxf(mrow[r], mx);
            const float alpha = __expf(mrow[r] - mnew);
            const float p0 = __expf(v0 - mnew);
            const float p1 = __expf(v1 - mnew);
            float rs = p0 + p1;
#pragma unroll
            for (int off = 8; off > 0; off >>= 1)
                rs += __shfl_xor(rs, off, 32);
            lrow[r] = lrow[r] * alpha + rs;
            mrow[r] = mnew;
#pragma unroll
            for (int db = 0; db < 4; ++db) acc[db][r] *= alpha;
            // stage P to LDS (C-layout -> memory row-major 16x32)
            Pl[wid][(r + 8 * half) * 32 + nIdx]      = (bf16)p0;
            Pl[wid][(r + 8 * half) * 32 + 16 + nIdx] = (bf16)p1;
        }

        // ---- reload P in A-layout from LDS; O += P V ----
        FragU aP;
        {
            const bf16* p = &Pl[wid][nIdx * 32 + half * 16];
            aP.h[0] = *(const v8bf*)p;
            aP.h[1] = *(const v8bf*)(p + 8);
        }
#pragma unroll
        for (int db = 0; db < 4; ++db) {
            // B fragment from vT: lane = d column, K = contiguous keys
            const bf16* vp = vbase + (size_t)(db * 16 + nIdx) * L_ + jb + half * 16;
            FragU bv;
            bv.h[0] = *(const v8bf*)vp;
            bv.h[1] = *(const v8bf*)(vp + 8);
            acc[db] = wmma_bf16(aP.v, bv.v, acc[db]);
        }
    }

    // ---- normalize and store (bf16 activations for out-projection GEMM) ----
#pragma unroll
    for (int db = 0; db < 4; ++db)
#pragma unroll
        for (int r = 0; r < 8; ++r) {
            const int row = rb + r + 8 * half;
            const float o = acc[db][r] / lrow[r];
            attnout[(baseBL + row) * 768 + h * 64 + db * 16 + nIdx] = (bf16)o;
        }
}

// ---------------------------------------------------------------------------
// Host orchestration
// ---------------------------------------------------------------------------
extern "C" void kernel_launch(void* const* d_in, const int* in_sizes, int n_in,
                              void* d_out, int out_size, void* d_ws, size_t ws_size,
                              hipStream_t stream) {
    const float* x      = (const float*)d_in[0];
    const unsigned char* amask = (const unsigned char*)d_in[1];
    const float* ln1_g  = (const float*)d_in[2];
    const float* ln1_b  = (const float*)d_in[3];
    const float* w_qkv  = (const float*)d_in[4];
    const float* b_qkv  = (const float*)d_in[5];
    const float* w_o    = (const float*)d_in[6];
    const float* b_o    = (const float*)d_in[7];
    const float* ln2_g  = (const float*)d_in[8];
    const float* ln2_b  = (const float*)d_in[9];
    const float* w1     = (const float*)d_in[10];
    const float* b1     = (const float*)d_in[11];
    const float* w2     = (const float*)d_in[12];
    const float* b2     = (const float*)d_in[13];

    char* ws = (char*)d_ws;
    size_t off = 0;
    auto alloc = [&](size_t bytes) -> void* {
        void* p = ws + off;
        off = (off + bytes + 255) & ~(size_t)255;
        return p;
    };

    bf16*  wqkvT  = (bf16*)alloc((size_t)(3 * D_) * D_ * 2);  // [3D x D]
    bf16*  woT    = (bf16*)alloc((size_t)D_ * D_ * 2);        // [D x D]
    bf16*  w1T    = (bf16*)alloc((size_t)I_ * D_ * 2);        // [I x D]
    bf16*  w2T    = (bf16*)alloc((size_t)D_ * I_ * 2);        // [D x I]
    bf16*  hbf    = (bf16*)alloc((size_t)M_ * D_ * 2);
    bf16*  qkvbf  = (bf16*)alloc((size_t)M_ * 3 * D_ * 2);
    bf16*  vTbuf  = (bf16*)alloc((size_t)B_ * H_ * HD_ * L_ * 2);
    bf16*  attnbf = (bf16*)alloc((size_t)M_ * D_ * 2);
    float* outatt = (float*)alloc((size_t)M_ * D_ * 4);
    bf16*  h2bf   = (bf16*)alloc((size_t)M_ * D_ * 2);
    bf16*  ff1bf  = (bf16*)alloc((size_t)M_ * I_ * 2);
    (void)ws_size; (void)in_sizes; (void)n_in; (void)out_size;

    // 1. weights -> bf16, transposed to [N x K]
    transpose_to_bf16<<<((size_t)D_ * 3 * D_ + 255) / 256, 256, 0, stream>>>(w_qkv, wqkvT, D_, 3 * D_);
    transpose_to_bf16<<<((size_t)D_ * D_ + 255) / 256, 256, 0, stream>>>(w_o, woT, D_, D_);
    transpose_to_bf16<<<((size_t)D_ * I_ + 255) / 256, 256, 0, stream>>>(w1, w1T, D_, I_);
    transpose_to_bf16<<<((size_t)I_ * D_ + 255) / 256, 256, 0, stream>>>(w2, w2T, I_, D_);

    // 2. h = LN1(x)  (bf16)
    ln_bf16_kernel<<<M_, 256, 0, stream>>>(x, ln1_g, ln1_b, hbf);

    // 3. qkv = h @ w_qkv + b_qkv  (bf16 out + V scattered transposed)
    gemm_wmma<<<dim3((3 * D_) / 256, M_ / 64), 256, 0, stream>>>(
        hbf, wqkvT, b_qkv, nullptr, nullptr, qkvbf, vTbuf, M_, 3 * D_, D_, 0);

    // 4. causal flash attention
    attn_kernel<<<(B_ * H_ * (L_ / 16)) / 8, 256, 0, stream>>>(qkvbf, vTbuf, amask, attnbf);

    // 5. out_attn = attn @ w_o + b_o + x  (f32)
    gemm_wmma<<<dim3(D_ / 256, M_ / 64), 256, 0, stream>>>(
        attnbf, woT, b_o, x, outatt, nullptr, nullptr, M_, D_, D_, 0);

    // 6. h2 = LN2(out_attn)  (bf16)
    ln_bf16_kernel<<<M_, 256, 0, stream>>>(outatt, ln2_g, ln2_b, h2bf);

    // 7. ff1 = gelu(h2 @ w1 + b1)  (bf16)
    gemm_wmma<<<dim3(I_ / 256, M_ / 64), 256, 0, stream>>>(
        h2bf, w1T, b1, nullptr, nullptr, ff1bf, nullptr, M_, I_, D_, 1);

    // 8. out = ff1 @ w2 + b2 + out_attn  (f32 -> d_out)
    gemm_wmma<<<dim3(D_ / 256, M_ / 64), 256, 0, stream>>>(
        ff1bf, w2T, b2, outatt, (float*)d_out, nullptr, nullptr, M_, D_, I_, 0);
}
